// GATLayer_3066606649846
// MI455X (gfx1250) — matrix operand order repaired and checked by reading, not compiled, not run on hardware
//
#include <hip/hip_runtime.h>
#include <hip/hip_bf16.h>

#define B_   4
#define L_   512
#define HID_ 512
#define H_   8
#define DH_  64
#define RS_  528   // padded LDS row stride (bf16 elems) for ctx_rel transposed tile

typedef __attribute__((ext_vector_type(16))) __bf16   v16bf;
typedef __attribute__((ext_vector_type(8)))  __bf16   v8bf;
typedef __attribute__((ext_vector_type(8)))  float    v8f;
typedef __attribute__((ext_vector_type(4)))  unsigned v4u;
typedef __attribute__((ext_vector_type(8)))  int      v8i;
typedef __attribute__((ext_vector_type(4)))  int      v4i;

__device__ __forceinline__ __bf16 f2bf(float f) { return (__bf16)f; }  // native v_cvt

__device__ __forceinline__ v16bf cat8(v8bf lo, v8bf hi) {
  return __builtin_shufflevector(lo, hi, 0,1,2,3,4,5,6,7,8,9,10,11,12,13,14,15);
}
__device__ __forceinline__ v16bf zero16() { v8bf z = {}; return cat8(z, z); }

__device__ __forceinline__ v8f wmma_bf16(v16bf a, v16bf b, v8f c) {
  return __builtin_amdgcn_wmma_f32_16x16x32_bf16(false, a, false, b, (short)0, c, false, false);
}

// A fragment (16x32 bf16) from a row-major bf16 row pointer (row = lane&15).
// This code's contiguous k order: elements 0..7 are cols kk+hi*16.., elements 8..15 are cols kk+hi*16+8..
// i.e. exactly two contiguous 16-byte loads per lane.
__device__ __forceinline__ v16bf load_a_bf16(const __bf16* row, int kk, int hi) {
  v8bf lo = *(const v8bf*)(row + kk + hi * 16);
  v8bf hh = *(const v8bf*)(row + kk + hi * 16 + 8);
  return cat8(lo, hh);
}
// B fragment: lane n = lane&15 owns column n; element e holds K = kk + e + 16*hi,
// so a K-contiguous source gives one 32-byte vector load (done inline below).

// ---------------------------------------------------------------------------
// K1: Q/K/V = hs @ W^T + b.  Q,K bf16 (B,H,L,DH); V bf16 transposed (B,H,DH,L)
// so the P@V B-operand is K-contiguous later.
// ---------------------------------------------------------------------------
__global__ void qkv_kernel(const float* __restrict__ hs,
                           const float* __restrict__ Wq, const float* __restrict__ bq,
                           const float* __restrict__ Wk, const float* __restrict__ bk,
                           const float* __restrict__ Wv, const float* __restrict__ bv,
                           __bf16* __restrict__ Qbf, __bf16* __restrict__ Kbf,
                           __bf16* __restrict__ Vt) {
  const int wave = threadIdx.x >> 5;
  const int lane = threadIdx.x & 31;
  const int hi   = lane >> 4;
  const int lm   = lane & 15;

  const int TILES = (B_ * L_ / 16) * (HID_ / 16);   // 4096 per matrix
  int t     = blockIdx.x * 8 + wave;
  int which = t / TILES;
  int rem   = t % TILES;
  int m0    = (rem / (HID_ / 16)) * 16;
  int n0    = (rem % (HID_ / 16)) * 16;

  const float* W    = (which == 0) ? Wq : (which == 1) ? Wk : Wv;
  const float* bias = (which == 0) ? bq : (which == 1) ? bk : bv;
  __bf16*      dst  = (which == 0) ? Qbf : (which == 1) ? Kbf : Vt;

  const float* arow = hs + (size_t)(m0 + lm) * HID_;
  const float* brow = W  + (size_t)(n0 + lm) * HID_ + hi * 16;

  v8f acc = {};
  for (int kk = 0; kk < HID_; kk += 32) {
    v8f alo = *(const v8f*)(arow + kk + hi * 16);
    v8f ahi = *(const v8f*)(arow + kk + hi * 16 + 8);
    v8f bl0 = *(const v8f*)(brow + kk);
    v8f bl1 = *(const v8f*)(brow + kk + 8);
    v16bf a, bm;
#pragma unroll
    for (int e = 0; e < 8; ++e) {
      a[e]      = f2bf(alo[e]);
      a[e + 8]  = f2bf(ahi[e]);
      bm[e]     = f2bf(bl0[e]);
      bm[e + 8] = f2bf(bl1[e]);
    }
    acc = wmma_bf16(a, bm, acc);
  }

  float bb = bias[n0 + lm];
  const bool isV = (which == 2);
#pragma unroll
  for (int r = 0; r < 8; ++r) {
    int gm = m0 + r + hi * 8;
    int f  = n0 + lm;
    int bi = gm / L_, l = gm % L_;
    int h  = f / DH_, d = f % DH_;
    __bf16 v = f2bf(acc[r] + bb);
    if (isV) dst[(((size_t)(bi * H_ + h) * DH_) + d) * L_ + l] = v;     // (B,H,DH,L)
    else     dst[(((size_t)(bi * H_ + h) * L_)  + l) * DH_ + d] = v;    // (B,H,L,DH)
  }
}

// ---------------------------------------------------------------------------
// K2: rel_scores[b,h,q,k] = sum_d Q[b,h,q,d] * rels[b,q,k,d]
// One block per (b,q); heads = M rows (8 used of 16). rels tile (512x64 f32,
// 128KB) staged in LDS via the Tensor Data Mover when available.
// ---------------------------------------------------------------------------
__global__ void relscore_kernel(const float* __restrict__ rels,
                                const __bf16* __restrict__ Qbf,
                                float* __restrict__ rsc) {
  extern __shared__ char smemraw[];
  float* Rf = (float*)smemraw;                        // [512][64] f32 (dyn LDS base)

  const int b = blockIdx.x / L_;
  const int q = blockIdx.x % L_;
  const int tid = threadIdx.x;
  const int wave = tid >> 5, lane = tid & 31, hi = lane >> 4, lm = lane & 15;

  const float* rsrc = rels + (size_t)(b * L_ + q) * (L_ * DH_);

#if __has_builtin(__builtin_amdgcn_tensor_load_to_lds)
  if (wave == 0) {
    // 2-D TDM descriptor: dim0 = DH (contig, stride DH), dim1 = L rows, 4B elems.
    unsigned long long ga = (unsigned long long)rsrc;
    v4u g0 = { 1u,                                        // count=1, no gather
               0u,                                        // lds_addr: dyn-LDS base
               (unsigned)ga,                              // global_addr[31:0]
               (unsigned)((ga >> 32) & 0x1FFFFFFull) | (2u << 30) }; // [56:32]|type=2
    v8i g1 = { (int)(2u << 16),                           // data_size = 4B
               (int)(DH_ << 16),                          // tensor_dim0 = 64
               (int)((unsigned)L_ << 16),                 // tensor_dim1 = 512
               (int)(DH_ << 16),                          // tile_dim0   = 64
               L_,                                        // tile_dim1   = 512
               DH_,                                       // tensor_dim0_stride = 64
               0, 0 };
    v4i gz = { 0, 0, 0, 0 };
#if defined(__clang_major__) && (__clang_major__ >= 23)
    v8i gz8 = { 0, 0, 0, 0, 0, 0, 0, 0 };
    __builtin_amdgcn_tensor_load_to_lds(g0, g1, gz, gz, gz8, 0);
#else
    __builtin_amdgcn_tensor_load_to_lds(g0, g1, gz, gz, 0);
#endif
    __builtin_amdgcn_s_wait_tensorcnt((short)0);
  }
#else
  for (int i = tid; i < L_ * DH_; i += 256) {
    __builtin_prefetch(rsrc + i + 4096, 0, 1);
    Rf[i] = rsrc[i];
  }
#endif
  __syncthreads();

  // A: 16x64, rows 0..7 = heads of Q[b,:,q,:], rows 8..15 zero
  const __bf16* qrow = Qbf + (((size_t)(b * H_ + (lm & 7)) * L_) + q) * DH_;
  const bool valid = (lm < H_);
  v16bf zv = zero16();
  v16bf a0 = valid ? load_a_bf16(qrow, 0, hi)  : zv;
  v16bf a1 = valid ? load_a_bf16(qrow, 32, hi) : zv;

  for (int i = 0; i < 4; ++i) {                        // 8 waves x 4 tiles = N=512
    int n0 = (wave * 4 + i) * 16;
    const float* bb = Rf + (n0 + lm) * DH_ + hi * 16;  // B(d,k) = R[k][d], K=d contig
    v8f c00 = *(const v8f*)(bb);
    v8f c01 = *(const v8f*)(bb + 8);
    v8f c10 = *(const v8f*)(bb + 32);
    v8f c11 = *(const v8f*)(bb + 40);
    v16bf b0, b1;
#pragma unroll
    for (int e = 0; e < 8; ++e) {
      b0[e] = f2bf(c00[e]); b0[e + 8] = f2bf(c01[e]);
      b1[e] = f2bf(c10[e]); b1[e + 8] = f2bf(c11[e]);
    }
    v8f acc = {};
    acc = wmma_bf16(a0, b0, acc);
    acc = wmma_bf16(a1, b1, acc);
    if (hi == 0) {
#pragma unroll
      for (int r = 0; r < H_; ++r)
        rsc[((size_t)((b * H_ + r) * L_) + q) * L_ + n0 + lm] = acc[r];
    }
  }
}

// ---------------------------------------------------------------------------
// K3: scores = Q@K^T + rel_scores, *0.125, + mask, row softmax, probs -> bf16
// ---------------------------------------------------------------------------
__global__ void score_softmax_kernel(const __bf16* __restrict__ Qbf,
                                     const __bf16* __restrict__ Kbf,
                                     const float* __restrict__ rsc,
                                     const float* __restrict__ mask,
                                     __bf16* __restrict__ probs) {
  __shared__ float S[16][L_];                          // 32KB
  const int qt = blockIdx.x % (L_ / 16);
  const int bh = blockIdx.x / (L_ / 16);
  const int b  = bh / H_;
  const int q0 = qt * 16;
  const int tid = threadIdx.x, wave = tid >> 5, lane = tid & 31, hi = lane >> 4, lm = lane & 15;

  const __bf16* qrow = Qbf + ((size_t)(bh * L_) + q0 + lm) * DH_;
  v16bf a0 = load_a_bf16(qrow, 0, hi);
  v16bf a1 = load_a_bf16(qrow, 32, hi);

  for (int i = 0; i < 4; ++i) {
    int n0 = (wave * 4 + i) * 16;
    const __bf16* kbase = Kbf + ((size_t)(bh * L_) + n0 + lm) * DH_ + hi * 16;
    v16bf b0 = *(const v16bf*)(kbase);                 // K=d contiguous
    v16bf b1 = *(const v16bf*)(kbase + 32);
    v8f acc = {};
    acc = wmma_bf16(a0, b0, acc);
    acc = wmma_bf16(a1, b1, acc);
#pragma unroll
    for (int r = 0; r < 8; ++r) {
      int M = r + hi * 8, col = n0 + lm;
      float s = (acc[r] + rsc[((size_t)(bh * L_) + q0 + M) * L_ + col]) * 0.125f
              + mask[((size_t)(b * L_) + q0 + M) * L_ + col];
      S[M][col] = s;
    }
  }
  __syncthreads();

#pragma unroll
  for (int rr = 0; rr < 2; ++rr) {
    int row = wave * 2 + rr;
    float m = -3.4e38f;
    for (int k = lane; k < L_; k += 32) m = fmaxf(m, S[row][k]);
#pragma unroll
    for (int off = 16; off > 0; off >>= 1) m = fmaxf(m, __shfl_xor(m, off, 32));
    float sum = 0.f;
    for (int k = lane; k < L_; k += 32) {
      float e = __expf(S[row][k] - m);
      S[row][k] = e;
      sum += e;
    }
#pragma unroll
    for (int off = 16; off > 0; off >>= 1) sum += __shfl_xor(sum, off, 32);
    float inv = 1.0f / sum;
    for (int k = lane; k < L_; k += 32)
      probs[((size_t)(bh * L_) + q0 + row) * L_ + k] = f2bf(S[row][k] * inv);
  }
}

// ---------------------------------------------------------------------------
// K4a: ctx_main = probs @ V  (V transposed (B,H,DH,L): B-operand contiguous)
// split-K across 8 waves + LDS reduction
// ---------------------------------------------------------------------------
__global__ void ctx_main_kernel(const __bf16* __restrict__ probs,
                                const __bf16* __restrict__ Vt,
                                float* __restrict__ out) {
  __shared__ float partial[4][16][16];
  const int qt = blockIdx.x % (L_ / 16);
  const int bh = blockIdx.x / (L_ / 16);
  const int b  = bh / H_, h = bh % H_;
  const int q0 = qt * 16;
  const int tid = threadIdx.x, wave = tid >> 5, lane = tid & 31, hi = lane >> 4, lm = lane & 15;
  const int nt = wave & 3, kh = wave >> 2;
  const int n0 = nt * 16;

  const __bf16* prow  = probs + ((size_t)(bh * L_) + q0 + lm) * L_;
  const __bf16* vbase = Vt + ((size_t)(bh * DH_) + n0 + lm) * L_;     // column d = n0+lm

  v8f acc = {};
  for (int c = 0; c < 8; ++c) {
    int kk = (kh * 8 + c) * 32;
    v16bf a  = load_a_bf16(prow, kk, hi);
    v16bf bm = *(const v16bf*)(vbase + kk + hi * 16);  // K=k contiguous
    acc = wmma_bf16(a, bm, acc);
  }
  if (kh == 1) {
#pragma unroll
    for (int r = 0; r < 8; ++r) partial[nt][r + hi * 8][lm] = acc[r];
  }
  __syncthreads();
  if (kh == 0) {
#pragma unroll
    for (int r = 0; r < 8; ++r) {
      int M = r + hi * 8;
      out[((size_t)(b * L_) + q0 + M) * HID_ + h * DH_ + n0 + lm]
          = acc[r] + partial[nt][M][lm];
    }
  }
}

// ---------------------------------------------------------------------------
// K4b: ctx += probs_allheads(8x512 padded) @ rels[b,q](512x64)
// rels tile staged TRANSPOSED in LDS (padded stride) so B-operand is contiguous.
// ---------------------------------------------------------------------------
__global__ void ctx_rel_kernel(const float* __restrict__ rels,
                               const __bf16* __restrict__ probs,
                               float* __restrict__ out) {
  extern __shared__ char smemraw[];
  __bf16* Rt = (__bf16*)smemraw;                       // [DH_][RS_], cols 0..511 used

  const int b = blockIdx.x / L_;
  const int q = blockIdx.x % L_;
  const int tid = threadIdx.x, wave = tid >> 5, lane = tid & 31, hi = lane >> 4, lm = lane & 15;

  const float* rsrc = rels + (size_t)(b * L_ + q) * (L_ * DH_);
  for (int i = tid; i < L_ * DH_; i += 256) {          // coalesced read, transpose
    __builtin_prefetch(rsrc + i + 4096, 0, 1);
    int k = i >> 6, d = i & 63;
    Rt[d * RS_ + k] = f2bf(rsrc[i]);
  }
  __syncthreads();

  if (wave >= 4) return;
  const int n0 = wave * 16;                            // d-tile
  const bool valid = (lm < H_);
  const __bf16* prow = probs + (((size_t)(b * H_ + (lm & 7)) * L_) + q) * L_;
  v16bf zv = zero16();

  v8f acc = {};
  for (int kk = 0; kk < L_; kk += 32) {
    v16bf a  = valid ? load_a_bf16(prow, kk, hi) : zv;
    v16bf bm = *(const v16bf*)(Rt + (n0 + lm) * RS_ + kk + hi * 16);  // K contiguous
    acc = wmma_bf16(a, bm, acc);
  }
  if (hi == 0) {
#pragma unroll
    for (int r = 0; r < H_; ++r) {
      size_t o = ((size_t)(b * L_) + q) * HID_ + r * DH_ + n0 + lm;
      out[o] += acc[r];                                // each element exactly once
    }
  }
}

// ---------------------------------------------------------------------------
extern "C" void kernel_launch(void* const* d_in, const int* in_sizes, int n_in,
                              void* d_out, int out_size, void* d_ws, size_t ws_size,
                              hipStream_t stream) {
  const float* hs    = (const float*)d_in[0];
  const float* heads = (const float*)d_in[1];
  const float* rels  = (const float*)d_in[2];
  const float* Wq    = (const float*)d_in[3];
  const float* bq    = (const float*)d_in[4];
  const float* Wk    = (const float*)d_in[5];
  const float* bk    = (const float*)d_in[6];
  const float* Wv    = (const float*)d_in[7];
  const float* bv    = (const float*)d_in[8];
  float* out = (float*)d_out;

  char* ws = (char*)d_ws;
  __bf16* Qbf   = (__bf16*)(ws);                 // 2MB
  __bf16* Kbf   = (__bf16*)(ws + (2u << 20));    // 2MB
  __bf16* Vt    = (__bf16*)(ws + (4u << 20));    // 2MB, transposed (B,H,DH,L)
  float*  rsc   = (float*) (ws + (6u << 20));    // 32MB
  __bf16* probs = (__bf16*)(ws + (38u << 20));   // 16MB

  qkv_kernel<<<1536, 256, 0, stream>>>(hs, Wq, bq, Wk, bk, Wv, bv, Qbf, Kbf, Vt);
  relscore_kernel<<<B_ * L_, 256, L_ * DH_ * 4, stream>>>(rels, Qbf, rsc);
  score_softmax_kernel<<<B_ * H_ * (L_ / 16), 256, 0, stream>>>(Qbf, Kbf, rsc, heads, probs);
  ctx_main_kernel<<<B_ * H_ * (L_ / 16), 256, 0, stream>>>(probs, Vt, out);
  ctx_rel_kernel<<<B_ * L_, 256, DH_ * RS_ * 2, stream>>>(rels, probs, out);
}
